// CharEmbedding_72129680769427
// MI455X (gfx1250) — matrix operand order repaired and checked by reading, not compiled, not run on hardware
//
#include <hip/hip_runtime.h>

typedef _Float16 f16;
typedef __attribute__((ext_vector_type(16))) _Float16 v16h;
typedef __attribute__((ext_vector_type(8)))  _Float16 v8h;
typedef __attribute__((ext_vector_type(4)))  _Float16 v4h;
typedef __attribute__((ext_vector_type(8)))  float    v8f;
typedef __attribute__((ext_vector_type(4)))  float    v4f;

#define HID   256
#define KDIM  768
#define ODIM  768
#define MIDN  12
#define BATCH 16
#define SEQ   2048
#define NTOK  (BATCH * SEQ)   // 32768

#define KS    32              // K per stage == one WMMA-K
#define A_STR 40              // halfs per A row (32 + 8 pad): aligned, bank-optimal
#define B_STR 40              // halfs per Bt row (k-major), same padding

// ---------------------------------------------------------------------------
// Kernel 1: midsum[u][c] = sum_j emb[mid[u][j]][c] * (mid!=0)   (30.7 MB, L2-resident)
// ---------------------------------------------------------------------------
__global__ __launch_bounds__(HID)
void midsum_kernel(const float* __restrict__ emb,
                   const int*   __restrict__ mid,
                   float*       __restrict__ midsum)
{
    __shared__ int sid[MIDN];
    const int u = blockIdx.x;
    const int c = threadIdx.x;
    if (c < MIDN) sid[c] = mid[u * MIDN + c];
    __syncthreads();
    float acc = 0.0f;
#pragma unroll
    for (int j = 0; j < MIDN; ++j) {
        const int id = sid[j];
        if (id != 0) acc += emb[(size_t)id * HID + c];
    }
    midsum[(size_t)u * HID + c] = acc;
}

// ---------------------------------------------------------------------------
// Kernel 2: one-time transpose+convert: wT16[n][k] = (f16) head_w[k][n]
// ---------------------------------------------------------------------------
__global__ __launch_bounds__(256)
void wt16_kernel(const float* __restrict__ w, f16* __restrict__ wt)
{
    __shared__ float t[64][65];
    const int n0 = (blockIdx.x % (ODIM / 64)) * 64;
    const int k0 = (blockIdx.x / (ODIM / 64)) * 64;
    const int tx = threadIdx.x & 63;
    const int ty = threadIdx.x >> 6;          // 0..3
#pragma unroll
    for (int r = 0; r < 16; ++r) {
        const int kr = ty * 16 + r;
        t[kr][tx] = w[(size_t)(k0 + kr) * ODIM + n0 + tx];
    }
    __syncthreads();
#pragma unroll
    for (int r = 0; r < 16; ++r) {
        const int nr = ty * 16 + r;
        wt[(size_t)(n0 + nr) * KDIM + k0 + tx] = (f16)t[tx][nr];
    }
}

// ---------------------------------------------------------------------------
// Kernel 3: zero the 2 pad rows per sequence
// ---------------------------------------------------------------------------
__global__ __launch_bounds__(256)
void pad_zero_kernel(float* __restrict__ out)
{
    const int b     = blockIdx.x >> 1;
    const int which = blockIdx.x & 1;
    const int row   = b * (SEQ + 2) + (which ? (SEQ + 1) : 0);
    float* p = out + (size_t)row * ODIM;
    for (int i = threadIdx.x; i < ODIM; i += blockDim.x) p[i] = 0.0f;
}

// ---------------------------------------------------------------------------
// Kernel 4: fused token-space GEMM with f16 WMMA 16x16x32.
//   Per block: 64 tokens x 256 out-cols; 8 waves in 2(M) x 4(N), wave = 32x64.
//   Outer loop over the 3 A-source segments (emb/first, midsum, emb/last) is
//   fully unrolled so the hot inner loop has a uniform source pointer and no
//   per-stage branching. All 12 fragment ds_loads issue before the 8 WMMAs.
// ---------------------------------------------------------------------------
__global__ __launch_bounds__(256)
void fused_gemm_kernel(const float* __restrict__ emb,
                       const float* __restrict__ head_b,
                       const int*   __restrict__ first,
                       const int*   __restrict__ last,
                       const int*   __restrict__ inv_i,
                       const float* __restrict__ midsum,
                       const f16*   __restrict__ wt16,
                       float*       __restrict__ out)
{
    __shared__ f16 Alds[64 * A_STR];          //  5.1 KB
    __shared__ f16 Blds[256 * B_STR];         // 20.5 KB
    __shared__ int s_u[64], s_f[64], s_l[64];

    const int tid = threadIdx.x;
    const int p0  = blockIdx.x * 64;          // first token of this tile
    const int n0  = blockIdx.y * 256;         // first output column

    if (tid < 64) {
        const int u = inv_i[p0 + tid];
        s_u[tid] = u;
        s_f[tid] = first[u];
        s_l[tid] = last[u];
    }
    __syncthreads();

    // ---- A staging assignment: thread -> (row, 8 consecutive k cols) ----
    const int arow = tid >> 2;                // 0..63
    const int acol = (tid & 3) * 8;           // 0,8,16,24
    const int a_u  = s_u[arow];
    const int a_f  = s_f[arow];
    const int a_l  = s_l[arow];
    const float* segsrc[3] = { emb    + (size_t)a_f * HID + acol,
                               midsum + (size_t)a_u * HID + acol,
                               emb    + (size_t)a_l * HID + acol };
    const float segmsk[3]  = { a_f ? 1.0f : 0.0f, 1.0f, a_l ? 1.0f : 0.0f };
    f16* const adst = &Alds[arow * A_STR + acol];

    // ---- B staging assignment: thread -> one column (32 halfs, contiguous) ----
    const f16* const bsrc_base = wt16 + (size_t)(n0 + tid) * KDIM;
    f16* const       bdst      = &Blds[tid * B_STR];

    // ---- wave / lane mapping ----
    const int lane = tid & 31;
    const int ln   = lane & 15;
    const int lh   = lane >> 4;
    const int wid  = tid >> 5;                // 8 waves: 2 (M) x 4 (N)
    const int m0w  = (wid & 1) * 32;
    const int n0w  = (wid >> 1) * 64;

    const f16* const Arow0 = &Alds[(m0w + ln) * A_STR];
    const f16* const Arow1 = &Alds[(m0w + 16 + ln) * A_STR];

    v8f acc[2][4] = {};                       // 32 x 64 f32 per wave

#pragma unroll
    for (int seg = 0; seg < 3; ++seg) {
        const float* const src_base = segsrc[seg];
        const float        msk      = segmsk[seg];
        const int          kseg     = seg * HID;

        for (int kk = 0; kk < HID; kk += KS) {
            const int k0 = kseg + kk;

            // ---- stage A (gathered, f32 -> f16) ----
            v4f a0 = *(const v4f*)(src_base + kk);
            v4f a1 = *(const v4f*)(src_base + kk + 4);
            a0 *= msk; a1 *= msk;
            const v4h h0 = __builtin_convertvector(a0, v4h);
            const v4h h1 = __builtin_convertvector(a1, v4h);
            *(v8h*)adst = __builtin_shufflevector(h0, h1, 0, 1, 2, 3, 4, 5, 6, 7);

            // ---- stage B (contiguous f16 copy from wT16) ----
            const f16* bs = bsrc_base + k0;
#pragma unroll
            for (int q = 0; q < 4; ++q)
                *(v8h*)(bdst + q * 8) = *(const v8h*)(bs + q * 8);

            __syncthreads();

            // ---- prefetch next K-stage sources while WMMAs run ----
            if (k0 + KS < KDIM) {
                __builtin_prefetch(src_base + ((kk + KS) & 255), 0, 3);
                __builtin_prefetch(bsrc_base + k0 + KS, 0, 3);
            }

            // ---- all fragment loads first (one clause / one wait) ----
            const v8h a0lo = *(const v8h*)(Arow0 + 8 * lh);
            const v8h a0hi = *(const v8h*)(Arow0 + 16 + 8 * lh);
            const v8h a1lo = *(const v8h*)(Arow1 + 8 * lh);
            const v8h a1hi = *(const v8h*)(Arow1 + 16 + 8 * lh);
            const v16h af0 = __builtin_shufflevector(a0lo, a0hi,
                0,1,2,3,4,5,6,7,8,9,10,11,12,13,14,15);
            const v16h af1 = __builtin_shufflevector(a1lo, a1hi,
                0,1,2,3,4,5,6,7,8,9,10,11,12,13,14,15);
            v16h bf[4];
#pragma unroll
            for (int j = 0; j < 4; ++j) {
                const f16* Brow = &Blds[(n0w + j * 16 + ln) * B_STR + 8 * lh];
                const v8h blo = *(const v8h*)(Brow);
                const v8h bhi = *(const v8h*)(Brow + 16);
                bf[j] = __builtin_shufflevector(blo, bhi,
                    0,1,2,3,4,5,6,7,8,9,10,11,12,13,14,15);
            }

            // ---- 8 back-to-back WMMAs ----
#pragma unroll
            for (int j = 0; j < 4; ++j) {
                acc[0][j] = __builtin_amdgcn_wmma_f32_16x16x32_f16(
                    false, af0, false, bf[j], (short)0, acc[0][j], false, false);
                acc[1][j] = __builtin_amdgcn_wmma_f32_16x16x32_f16(
                    false, af1, false, bf[j], (short)0, acc[1][j], false, false);
            }
            __syncthreads();
        }
    }

    // ---- epilogue: bias + scatter through pad mapping orow = p + 2*(p/SEQ) + 1 ----
#pragma unroll
    for (int j = 0; j < 4; ++j) {
        const int c    = n0 + n0w + j * 16 + ln;
        const float bv = head_b[c];
#pragma unroll
        for (int tm = 0; tm < 2; ++tm) {
            const int pbase = p0 + m0w + tm * 16;
#pragma unroll
            for (int v = 0; v < 8; ++v) {
                const int p    = pbase + v + 8 * lh;    // token index
                const int orow = p + 2 * (p >> 11) + 1; // SEQ = 2048 = 1<<11
                out[(size_t)orow * ODIM + c] = acc[tm][j][v] + bv;
            }
        }
    }
}

// ---------------------------------------------------------------------------
extern "C" void kernel_launch(void* const* d_in, const int* in_sizes, int n_in,
                              void* d_out, int out_size, void* d_ws, size_t ws_size,
                              hipStream_t stream)
{
    const float* emb    = (const float*)d_in[0];   // [4000, 256]
    const float* head_w = (const float*)d_in[1];   // [768, 768]
    const float* head_b = (const float*)d_in[2];   // [768]
    const int*   first  = (const int*)d_in[3];     // [U]
    const int*   mid    = (const int*)d_in[4];     // [U, 12]
    const int*   last   = (const int*)d_in[5];     // [U]
    const int*   inv_i  = (const int*)d_in[6];     // [32768]
    const int    U      = in_sizes[3];

    float* midsum = (float*)d_ws;                              // U*256 f32 = 30.7 MB
    f16*   wt16   = (f16*)((char*)d_ws + (size_t)U * HID * 4); // 768*768 f16 = 1.18 MB
    float* out    = (float*)d_out;                             // [16, 2050, 768] f32

    midsum_kernel<<<U, HID, 0, stream>>>(emb, mid, midsum);
    wt16_kernel<<<(ODIM / 64) * (KDIM / 64), 256, 0, stream>>>(head_w, wt16);
    pad_zero_kernel<<<BATCH * 2, 256, 0, stream>>>(out);
    fused_gemm_kernel<<<dim3(NTOK / 64, ODIM / 256), 256, 0, stream>>>(
        emb, head_b, first, last, inv_i, midsum, wt16, out);
}
